// Res_GN_20779051778390
// MI455X (gfx1250) — compile-verified, matrix operand order checked by reading, not run on hardware
//
#include <hip/hip_runtime.h>
#include <hip/hip_bf16.h>

// Problem constants (from reference)
#define T_ 4
#define N_ 20000
#define E_ 320000
// F_IN=12, EH=32, NH=64, GH=64, OUT=1, EB_IN=384, NB_IN=256, GB_IN=160

typedef __bf16 v16bf __attribute__((ext_vector_type(16)));
typedef float  v8f   __attribute__((ext_vector_type(8)));

__device__ __forceinline__ v8f wmma_bf16(v16bf a, v16bf b, v8f c) {
    // D = A(16x32 bf16) * B(32x16 bf16) + C(16x16 f32)
    return __builtin_amdgcn_wmma_f32_16x16x32_bf16(false, a, false, b, (short)0, c,
                                                   false, false);
}

// Build A fragment from one fp32 source row (chunk of 32 K-values).
// lane<16 : K {0..7,16..23}; lane>=16 : K {8..15,24..31}  (klo/khi precomputed)
__device__ __forceinline__ v16bf afrag(const float* __restrict__ src, int klo, int khi) {
    v16bf a;
#pragma unroll
    for (int j = 0; j < 8; ++j) {
        a[j]     = (__bf16)src[klo + j];
        a[8 + j] = (__bf16)src[khi + j];
    }
    return a;
}

__device__ __forceinline__ v16bf zfrag() {
    v16bf a;
#pragma unroll
    for (int j = 0; j < 16; ++j) a[j] = (__bf16)0.0f;
    return a;
}

// Cooperatively build B fragments (fragment-ordered) in LDS from fp32 weights.
// Slot layout: sB[(chunk*NT + nt)*32 + lane] = 16 bf16 (32 bytes).
// For lane: column n = (lane&15) + 16*nt, K-halves klo/khi per WMMA A/B layout.
__device__ __forceinline__ void fill_Bfrags(v16bf* sB, const float* __restrict__ W,
                                            int nchunks, int ntile, int ncols) {
    const int total = nchunks * ntile * 32;
    for (int idx = threadIdx.x; idx < total; idx += blockDim.x) {
        const int l  = idx & 31;
        const int nt = (idx >> 5) % ntile;
        const int c  = idx / (32 * ntile);
        const int kl = (l < 16) ? 0 : 8;
        const int kh = kl + 16;
        const int n  = (l & 15) + 16 * nt;
        v16bf b;
#pragma unroll
        for (int j = 0; j < 8; ++j) {
            b[j]     = (__bf16)W[(32 * c + kl + j) * ncols + n];
            b[8 + j] = (__bf16)W[(32 * c + kh + j) * ncols + n];
        }
        sB[idx] = b;
    }
}

// ---------------------------------------------------------------------------
// Edge block: e_new = relu(cat[e_in,h_e,x[s],h_x[s],x[r],h_x[r],g] @ W_eb + b)
// Fused: WMMA GEMM + store + scatter-add to sent/recv + e-mean partials.
// B fragments staged in LDS (24 KB) -> low VGPR count, high occupancy.
// ---------------------------------------------------------------------------
__global__ void __launch_bounds__(256)
edge_block_kernel(const float* __restrict__ e_in,   // [E,32]
                  const float* __restrict__ h_e,    // [E,32] or null (t==0)
                  const float* __restrict__ x_in,   // [N,64]
                  const float* __restrict__ h_x,    // [N,64] or null (t==0)
                  const float* __restrict__ g_in,   // [64]
                  const float* __restrict__ W,      // [384][32] fp32
                  const float* __restrict__ bias,   // [32]
                  const int* __restrict__ sidx,
                  const int* __restrict__ ridx,
                  float* __restrict__ e_out,        // [E,32]
                  float* __restrict__ sent,         // [N,32]
                  float* __restrict__ recv,         // [N,32]
                  float* __restrict__ esum)         // [32]
{
    __shared__ v16bf sB[12 * 2 * 32];                // 24 KB
    fill_Bfrags(sB, W, 12, 2, 32);
    __syncthreads();

    const int lane = threadIdx.x & 31;
    const int wid  = blockIdx.x * (blockDim.x >> 5) + (threadIdx.x >> 5);
    const int nw   = gridDim.x * (blockDim.x >> 5);
    const int lh   = lane & 15;
    const int klo  = (lane < 16) ? 0 : 8;
    const int khi  = klo + 16;

    const v16bf gA0 = afrag(g_in, klo, khi);
    const v16bf gA1 = afrag(g_in + 32, klo, khi);
    const float b0 = bias[lh], b1 = bias[lh + 16];

#define EB_MM(c, A)                                                       \
    do {                                                                  \
        acc0 = wmma_bf16((A), sB[((c) * 2 + 0) * 32 + lane], acc0);       \
        acc1 = wmma_bf16((A), sB[((c) * 2 + 1) * 32 + lane], acc1);       \
    } while (0)

    float em0 = 0.f, em1 = 0.f;
    const int ntiles = E_ / 16;
    for (int tile = wid; tile < ntiles; tile += nw) {
        const int base = tile * 16;
        const int row  = base + lh;
        const int s = sidx[row], r = ridx[row];

        v8f acc0 = {0.f,0.f,0.f,0.f,0.f,0.f,0.f,0.f};
        v8f acc1 = {0.f,0.f,0.f,0.f,0.f,0.f,0.f,0.f};
        v16bf a;

        a = afrag(e_in + (size_t)row * 32, klo, khi);
        EB_MM(0, a);
        if (h_e) {  // wave-uniform branch: EXEC stays full for WMMA
            a = afrag(h_e + (size_t)row * 32, klo, khi);
            EB_MM(1, a);
        }
        const float* xs = x_in + (size_t)s * 64;
        a = afrag(xs, klo, khi);       EB_MM(2, a);
        a = afrag(xs + 32, klo, khi);  EB_MM(3, a);
        if (h_x) {
            const float* hs = h_x + (size_t)s * 64;
            a = afrag(hs, klo, khi);       EB_MM(4, a);
            a = afrag(hs + 32, klo, khi);  EB_MM(5, a);
        }
        const float* xr = x_in + (size_t)r * 64;
        a = afrag(xr, klo, khi);       EB_MM(6, a);
        a = afrag(xr + 32, klo, khi);  EB_MM(7, a);
        if (h_x) {
            const float* hr = h_x + (size_t)r * 64;
            a = afrag(hr, klo, khi);       EB_MM(8, a);
            a = afrag(hr + 32, klo, khi);  EB_MM(9, a);
        }
        EB_MM(10, gA0);
        EB_MM(11, gA1);

        // Epilogue: D layout -> lane holds column n=lh(+16), rows m0..m0+7
        const int m0 = (lane < 16) ? 0 : 8;
#pragma unroll
        for (int v = 0; v < 8; ++v) {
            const int edge = base + m0 + v;
            float e0 = acc0[v] + b0; e0 = e0 > 0.f ? e0 : 0.f;
            float e1 = acc1[v] + b1; e1 = e1 > 0.f ? e1 : 0.f;
            e_out[(size_t)edge * 32 + lh]      = e0;
            e_out[(size_t)edge * 32 + lh + 16] = e1;
            const int sm = sidx[edge], rm = ridx[edge];
            atomicAdd(&sent[(size_t)sm * 32 + lh],      e0);
            atomicAdd(&sent[(size_t)sm * 32 + lh + 16], e1);
            atomicAdd(&recv[(size_t)rm * 32 + lh],      e0);
            atomicAdd(&recv[(size_t)rm * 32 + lh + 16], e1);
            em0 += e0; em1 += e1;
        }
    }
#undef EB_MM
    atomicAdd(&esum[lh],      em0);
    atomicAdd(&esum[lh + 16], em1);
}

// ---------------------------------------------------------------------------
// Node block: x_new = relu(cat[x_in,h_x,sent,recv,g] @ W_nb + b)  (K=256,N=64)
// Also emits t_der = x_new - h_x when td_out != null, and x-mean partials.
// B fragments staged in LDS (32 KB).
// ---------------------------------------------------------------------------
__global__ void __launch_bounds__(256)
node_block_kernel(const float* __restrict__ x_in,
                  const float* __restrict__ h_x,    // null at t==0
                  const float* __restrict__ sent,
                  const float* __restrict__ recv,
                  const float* __restrict__ g_in,
                  const float* __restrict__ W,      // [256][64] fp32
                  const float* __restrict__ bias,   // [64]
                  float* __restrict__ x_out,        // [N,64]
                  float* __restrict__ xsum,         // [64]
                  float* __restrict__ td_out)       // [N,64] or null
{
    __shared__ v16bf sB[8 * 4 * 32];                 // 32 KB
    fill_Bfrags(sB, W, 8, 4, 64);
    __syncthreads();

    const int lane = threadIdx.x & 31;
    const int wid  = blockIdx.x * (blockDim.x >> 5) + (threadIdx.x >> 5);
    const int nw   = gridDim.x * (blockDim.x >> 5);
    const int lh   = lane & 15;
    const int klo  = (lane < 16) ? 0 : 8;
    const int khi  = klo + 16;

    const v16bf g0f = afrag(g_in, klo, khi);
    const v16bf g1f = afrag(g_in + 32, klo, khi);

    float xm[4] = {0.f, 0.f, 0.f, 0.f};
    const int ntiles = N_ / 16;
    for (int tile = wid; tile < ntiles; tile += nw) {
        const int base = tile * 16;
        const int row  = base + lh;

        v16bf a[8];
        const float* xp = x_in + (size_t)row * 64;
        a[0] = afrag(xp, klo, khi);
        a[1] = afrag(xp + 32, klo, khi);
        if (h_x) {
            const float* hp = h_x + (size_t)row * 64;
            a[2] = afrag(hp, klo, khi);
            a[3] = afrag(hp + 32, klo, khi);
        } else { a[2] = zfrag(); a[3] = zfrag(); }
        a[4] = afrag(sent + (size_t)row * 32, klo, khi);
        a[5] = afrag(recv + (size_t)row * 32, klo, khi);
        a[6] = g0f; a[7] = g1f;

        v8f acc[4];
#pragma unroll
        for (int nt = 0; nt < 4; ++nt) acc[nt] = (v8f){0.f,0.f,0.f,0.f,0.f,0.f,0.f,0.f};
#pragma unroll
        for (int c = 0; c < 8; ++c)
#pragma unroll
            for (int nt = 0; nt < 4; ++nt)
                acc[nt] = wmma_bf16(a[c], sB[(c * 4 + nt) * 32 + lane], acc[nt]);

        const int m0 = (lane < 16) ? 0 : 8;
#pragma unroll
        for (int nt = 0; nt < 4; ++nt) {
            const int n = lh + 16 * nt;
            const float bb = bias[n];
#pragma unroll
            for (int v = 0; v < 8; ++v) {
                const int node = base + m0 + v;
                float x = acc[nt][v] + bb; x = x > 0.f ? x : 0.f;
                x_out[(size_t)node * 64 + n] = x;
                xm[nt] += x;
                if (td_out) {
                    const float hp = h_x ? h_x[(size_t)node * 64 + n] : 0.f;
                    td_out[(size_t)node * 64 + n] = x - hp;
                }
            }
        }
    }
#pragma unroll
    for (int nt = 0; nt < 4; ++nt) atomicAdd(&xsum[lh + 16 * nt], xm[nt]);
}

// ---------------------------------------------------------------------------
// Global block: g_new = relu([xmean, emean, h_g] @ W_gb + b)   (1x160 @ 160x64)
// ---------------------------------------------------------------------------
__global__ void global_block_kernel(const float* __restrict__ xsum,
                                    const float* __restrict__ esum,
                                    const float* __restrict__ g_in,
                                    const float* __restrict__ Wg,
                                    const float* __restrict__ bg,
                                    float* __restrict__ g_out,
                                    float* __restrict__ g_save)  // nullable
{
    __shared__ float gc[160];
    const int j = threadIdx.x;  // 64 threads
    gc[j] = xsum[j] * (1.0f / (float)N_);
    if (j < 32) gc[64 + j] = esum[j] * (1.0f / (float)E_);
    gc[96 + j] = g_in[j];
    __syncthreads();
    float acc = bg[j];
    for (int k = 0; k < 160; ++k) acc += gc[k] * Wg[k * 64 + j];
    acc = acc > 0.f ? acc : 0.f;
    g_out[j] = acc;
    if (g_save) g_save[j] = acc;
}

// ---------------------------------------------------------------------------
// Spatial derivative: sd[r] += coeff * spL[e] * (x[s] - x[r])   (scatter by r)
// ---------------------------------------------------------------------------
__global__ void sder_kernel(const float* __restrict__ x_new,
                            const int* __restrict__ sidx,
                            const int* __restrict__ ridx,
                            const float* __restrict__ spL,
                            const float* __restrict__ coeff,
                            float* __restrict__ sd)
{
    const int e = blockIdx.x * blockDim.x + threadIdx.x;
    if (e >= E_) return;
    const int s = sidx[e], r = ridx[e];
    const float w = coeff[0] * spL[e];
    const float4* xs = (const float4*)(x_new + (size_t)s * 64);
    const float4* xr = (const float4*)(x_new + (size_t)r * 64);
    float* o = sd + (size_t)r * 64;
#pragma unroll
    for (int i = 0; i < 16; ++i) {
        const float4 a = xs[i], b = xr[i];
        atomicAdd(&o[4 * i + 0], w * (a.x - b.x));
        atomicAdd(&o[4 * i + 1], w * (a.y - b.y));
        atomicAdd(&o[4 * i + 2], w * (a.z - b.z));
        atomicAdd(&o[4 * i + 3], w * (a.w - b.w));
    }
}

// ---------------------------------------------------------------------------
// Encoders / small kernels
// ---------------------------------------------------------------------------
__global__ void node_enc_kernel(const float* __restrict__ na, const float* __restrict__ W,
                                const float* __restrict__ b, float* __restrict__ xe)
{
    const int idx = blockIdx.x * blockDim.x + threadIdx.x;
    if (idx >= T_ * N_ * 64) return;
    const int col = idx & 63;
    const int row = idx >> 6;
    const float* p = na + (size_t)row * 12;
    float acc = b[col];
#pragma unroll
    for (int k = 0; k < 12; ++k) acc += p[k] * W[k * 64 + col];
    xe[idx] = acc > 0.f ? acc : 0.f;
}

__global__ void edge_enc_kernel(const float* __restrict__ ea, const float* __restrict__ W,
                                const float* __restrict__ b, float* __restrict__ ee)
{
    const long idx = (long)blockIdx.x * blockDim.x + threadIdx.x;
    if (idx >= (long)T_ * E_ * 32) return;
    const int col = (int)(idx & 31);
    const float v = ea[idx >> 5];
    const float acc = v * W[col] + b[col];
    ee[idx] = acc > 0.f ? acc : 0.f;
}

__global__ void resid_kernel(const float* __restrict__ a, const float* __restrict__ b,
                             float* __restrict__ o, int n)
{
    const int i = blockIdx.x * blockDim.x + threadIdx.x;
    if (i < n) o[i] = a[i] + b[i];
}

__global__ void copy_kernel(const float* __restrict__ src, float* __restrict__ dst, int n)
{
    const int i = blockIdx.x * blockDim.x + threadIdx.x;
    if (i < n) dst[i] = src[i];
}

// ---------------------------------------------------------------------------
// Decoder: out = relu((xs2+xres) @ W1 + b1) @ W2 + b2    (64x64 then 64x1)
// ---------------------------------------------------------------------------
__global__ void __launch_bounds__(256)
decoder_kernel(const float* __restrict__ xs2, const float* __restrict__ xres,
               const float* __restrict__ W1, const float* __restrict__ b1,
               const float* __restrict__ W2, const float* __restrict__ b2,
               float* __restrict__ out)
{
    __shared__ float sW1[64 * 64];
    __shared__ float sW2[64], sb1[64];
    for (int i = threadIdx.x; i < 4096; i += blockDim.x) sW1[i] = W1[i];
    if (threadIdx.x < 64) { sW2[threadIdx.x] = W2[threadIdx.x]; sb1[threadIdx.x] = b1[threadIdx.x]; }
    __syncthreads();
    const int row = blockIdx.x * blockDim.x + threadIdx.x;
    if (row >= T_ * N_) return;
    float xf[64];
    const float* p1 = xs2 + (size_t)row * 64;
    const float* p2 = xres + (size_t)row * 64;
#pragma unroll
    for (int k = 0; k < 64; ++k) xf[k] = p1[k] + p2[k];
    float o = b2[0];
    for (int j = 0; j < 64; ++j) {
        float h = sb1[j];
#pragma unroll 16
        for (int k = 0; k < 64; ++k) h += xf[k] * sW1[k * 64 + j];
        h = h > 0.f ? h : 0.f;
        o += h * sW2[j];
    }
    out[row] = o;
}

// ---------------------------------------------------------------------------
// Host orchestration
// ---------------------------------------------------------------------------
extern "C" void kernel_launch(void* const* d_in, const int* in_sizes, int n_in,
                              void* d_out, int out_size, void* d_ws, size_t ws_size,
                              hipStream_t stream)
{
    const float* node_attr   = (const float*)d_in[0];
    const float* edge_attr   = (const float*)d_in[1];
    const int*   edge_index  = (const int*)d_in[2];
    const float* sp_L_vals   = (const float*)d_in[3];
    const float* global_attr = (const float*)d_in[4];
    const float* coeff       = (const float*)d_in[5];
    const float* W_ee  = (const float*)d_in[6];
    const float* b_ee  = (const float*)d_in[7];
    const float* W_ne  = (const float*)d_in[8];
    const float* b_ne  = (const float*)d_in[9];
    const float* W_eb  = (const float*)d_in[10];
    const float* b_eb  = (const float*)d_in[11];
    const float* W_nb  = (const float*)d_in[12];
    const float* b_nb  = (const float*)d_in[13];
    const float* W_gb  = (const float*)d_in[14];
    const float* b_gb  = (const float*)d_in[15];
    const float* W_nd1 = (const float*)d_in[16];
    const float* b_nd1 = (const float*)d_in[17];
    const float* W_nd2 = (const float*)d_in[18];
    const float* b_nd2 = (const float*)d_in[19];

    const int* sidx = edge_index;
    const int* ridx = edge_index + E_;

    float* out_nodes = (float*)d_out;                       // [T*N]
    float* out_td    = out_nodes + (size_t)T_ * N_;         // [T*N*64]
    float* out_sd    = out_td + (size_t)T_ * N_ * 64;       // [T*N*64]

    float* ws = (float*)d_ws;
    size_t off = 0;
    auto take = [&](size_t n) { float* p = ws + off; off += (n + 63) & ~(size_t)63; return p; };
    float* x_enc = take((size_t)T_ * N_ * 64);   // node-encoder output
    float* es1   = take((size_t)T_ * E_ * 32);   // layer-1 edge states (all t)
    float* es2   = take((size_t)T_ * E_ * 32);   // edge-encoder output, reused as layer-2 edge states
    float* xs1   = take((size_t)T_ * N_ * 64);   // layer-1 node states
    float* xres  = take((size_t)T_ * N_ * 64);   // residual (layer-2 input)
    float* xs2   = take((size_t)T_ * N_ * 64);   // layer-2 node states
    float* sent  = take((size_t)N_ * 32);
    float* recv  = take((size_t)N_ * 32);
    float* esum  = take(64);
    float* xsum  = take(64);
    float* gA    = take(64);
    float* gB    = take(64);
    float* gS    = take(64);

    // Zero the s_der scatter target (output region) every launch.
    hipMemsetAsync(out_sd, 0, (size_t)T_ * N_ * 64 * sizeof(float), stream);

    // Encoders + initial global state.
    node_enc_kernel<<<(T_ * N_ * 64 + 255) / 256, 256, 0, stream>>>(node_attr, W_ne, b_ne, x_enc);
    edge_enc_kernel<<<(int)(((long)T_ * E_ * 32 + 255) / 256), 256, 0, stream>>>(edge_attr, W_ee, b_ee, es2);
    copy_kernel<<<1, 64, 0, stream>>>(global_attr, gA, 64);

    // sent|recv|esum|xsum are contiguous: one memset per step clears all four.
    const size_t zbytes = ((size_t)2 * N_ * 32 + 128) * sizeof(float);

    for (int layer = 0; layer < 2; ++layer) {
        const float* xT = (layer == 0) ? x_enc : xres;  // per-step node inputs
        const float* eT = (layer == 0) ? es2 : es1;     // per-step edge inputs
        float* eo = (layer == 0) ? es1 : es2;           // edge outputs (also next h_e)
        float* xo = (layer == 0) ? xs1 : xs2;           // node outputs (also next h_x)
        const bool last = (layer == 1);

        for (int t = 0; t < T_; ++t) {
            hipMemsetAsync(sent, 0, zbytes, stream);
            float* gin  = (t & 1) ? gB : gA;
            float* gout = (t & 1) ? gA : gB;

            edge_block_kernel<<<320, 256, 0, stream>>>(
                eT + (size_t)t * E_ * 32,
                t ? eo + (size_t)(t - 1) * E_ * 32 : nullptr,
                xT + (size_t)t * N_ * 64,
                t ? xo + (size_t)(t - 1) * N_ * 64 : nullptr,
                gin, W_eb, b_eb, sidx, ridx,
                eo + (size_t)t * E_ * 32, sent, recv, esum);

            node_block_kernel<<<160, 256, 0, stream>>>(
                xT + (size_t)t * N_ * 64,
                t ? xo + (size_t)(t - 1) * N_ * 64 : nullptr,
                sent, recv, gin, W_nb, b_nb,
                xo + (size_t)t * N_ * 64, xsum,
                last ? out_td + (size_t)t * N_ * 64 : nullptr);

            global_block_kernel<<<1, 64, 0, stream>>>(
                xsum, esum, gin, W_gb, b_gb, gout,
                (layer == 0 && t == 0) ? gS : nullptr);

            if (last)
                sder_kernel<<<(E_ + 255) / 256, 256, 0, stream>>>(
                    xo + (size_t)t * N_ * 64, sidx, ridx, sp_L_vals, coeff,
                    out_sd + (size_t)t * N_ * 64);
        }

        if (layer == 0) {
            resid_kernel<<<(T_ * N_ * 64 + 255) / 256, 256, 0, stream>>>(xs1, x_enc, xres, T_ * N_ * 64);
            copy_kernel<<<1, 64, 0, stream>>>(gS, gA, 64);  // layer-2 g0 = gs[0] of layer 1
        }
    }

    decoder_kernel<<<(T_ * N_ + 255) / 256, 256, 0, stream>>>(
        xs2, xres, W_nd1, b_nd1, W_nd2, b_nd2, out_nodes);
}
